// SwinTransformerBlock_8899172237876
// MI455X (gfx1250) — matrix operand order repaired and checked by reading, not compiled, not run on hardware
//
#include <hip/hip_runtime.h>
#include <hip/hip_bf16.h>
#include <cstdint>

// ---------- WMMA types ----------
typedef __attribute__((ext_vector_type(16))) _Float16 v16h;
typedef __attribute__((ext_vector_type(8)))  float    v8f;
typedef __attribute__((ext_vector_type(4)))  unsigned int u32x4;
typedef __attribute__((ext_vector_type(8)))  int      i32x8;
typedef __attribute__((ext_vector_type(4)))  int      i32x4;

struct u128 { unsigned int x, y, z, w; };         // POD 16B block
union AFrag { v16h v; u128 u[2]; };               // 32B fragment (8 VGPRs)

#define NHEAD 6
#define HDIM  32
#define CDIM  192

// ----------------------------------------------------------------------------
// TDM: DMA a 64-row x 32-half tile (row stride = strideElems halves) from
// global memory into LDS at lds_addr. D# per CDNA5 ISA ch.8 (2-D tile,
// data_size=2B, groups 2/3 zero). Issued by one wave; completion via TENSORcnt.
// ----------------------------------------------------------------------------
__device__ __forceinline__ void tdm_load_tile_64x32(unsigned long long gaddr,
                                                    unsigned lds_addr,
                                                    int strideElems) {
    u32x4 g0;
    g0.x = 1u;                                       // count=1, is_restore=0
    g0.y = lds_addr;                                 // LDS byte address
    g0.z = (unsigned)(gaddr & 0xffffffffull);        // global_addr[31:0]
    g0.w = (unsigned)(gaddr >> 32) | 0x80000000u;    // addr[56:32] | type=2
    i32x8 g1;
    g1[0] = 0x00010000;                              // wg_mask=0, data_size=2B
    g1[1] = (int)(32u << 16);                        // tensor_dim0 = 32 (lo16)
    g1[2] = (int)(64u << 16);                        // dim0 hi=0 | tensor_dim1=64 (lo16)
    g1[3] = (int)(32u << 16);                        // dim1 hi=0 | tile_dim0=32
    g1[4] = 64;                                      // tile_dim1=64, tile_dim2=0
    g1[5] = strideElems;                             // tensor_dim0_stride[31:0]
    g1[6] = 0;                                       // stride0 hi | stride1 lo
    g1[7] = 0;                                       // stride1 hi
    i32x4 z4 = (i32x4)0;
#if defined(__clang_major__) && (__clang_major__ >= 23)
    i32x8 z8 = (i32x8)0;
    __builtin_amdgcn_tensor_load_to_lds(g0, g1, z4, z4, z8, 0);
#else
    __builtin_amdgcn_tensor_load_to_lds(g0, g1, z4, z4, 0);
#endif
}

// ----------------------------------------------------------------------------
// f32 -> f16 convert (weights)
// ----------------------------------------------------------------------------
__global__ void cvt_f16_kernel(const float* __restrict__ in, _Float16* __restrict__ out, int n) {
    int i = blockIdx.x * blockDim.x + threadIdx.x;
    if (i < n) out[i] = (_Float16)in[i];
}

// ----------------------------------------------------------------------------
// LayerNorm over C=192, one wave per token (wave32), 8 tokens / 256-thread block
// ----------------------------------------------------------------------------
__global__ __launch_bounds__(256)
void ln_kernel(const float* __restrict__ x, const float* __restrict__ w,
               const float* __restrict__ b, _Float16* __restrict__ out, int Mtok) {
    int tok  = blockIdx.x * 8 + (threadIdx.x >> 5);
    int lane = threadIdx.x & 31;
    if (tok >= Mtok) return;
    const float* row = x + (size_t)tok * CDIM;
    float v[6], s = 0.f, sq = 0.f;
#pragma unroll
    for (int j = 0; j < 6; ++j) { v[j] = row[lane + j * 32]; s += v[j]; sq += v[j] * v[j]; }
#pragma unroll
    for (int off = 16; off > 0; off >>= 1) { s += __shfl_xor(s, off); sq += __shfl_xor(sq, off); }
    float mean = s * (1.0f / 192.0f);
    float var  = sq * (1.0f / 192.0f) - mean * mean;
    float rs   = rsqrtf(var + 1e-5f);
    _Float16* orow = out + (size_t)tok * CDIM;
#pragma unroll
    for (int j = 0; j < 6; ++j) {
        int i = lane + j * 32;
        orow[i] = (_Float16)((v[j] - mean) * rs * w[i] + b[i]);
    }
}

// ----------------------------------------------------------------------------
// WMMA GEMM: C[M,N] = A[M,K](f16,row) * B[K,N](f16,row) + bias, fused epilogue.
// Block 192 thr = 6 waves, macro tile 64x96, wave tile 32x48 (2x3 wave grid, wave cols 32 apart).
// A tile staged by the Tensor Data Mover (double buffered, TENSORcnt);
// B tile staged transposed by VALU (weights are L2-resident).
// mode 0: f16 out          (QKV)
// mode 1: f32 out + resid  (proj+shortcut, MLP2+residual)
// mode 2: f16 out, GELU    (MLP1)
// ----------------------------------------------------------------------------
__global__ __launch_bounds__(192)
void gemm_wmma_kernel(const _Float16* __restrict__ A, const _Float16* __restrict__ Bw,
                      const float* __restrict__ bias, const float* __restrict__ resid,
                      float* __restrict__ outF, _Float16* __restrict__ outH,
                      int M, int K, int N, int mode) {
    __shared__ __align__(16) _Float16 As[2][64 * 32];   // [m][k] stride 32 (TDM target)
    __shared__ __align__(16) _Float16 Bs[2][96 * 48];   // [n][k] stride 48 (transposed)

    const int tid  = threadIdx.x;
    const int lane = tid & 31;
    const int wv   = tid >> 5;
    const int m0   = blockIdx.y * 64;
    const int n0   = blockIdx.x * 96;
    const int wrow = (wv / 3) * 32;
    const int wcol = (wv % 3) * 32;

    const unsigned long long abase = (unsigned long long)(uintptr_t)A +
                                     2ull * ((size_t)m0 * (size_t)K);
    const unsigned ldsA0 = (unsigned)(uintptr_t)(const void*)&As[0][0];
    const unsigned ldsA1 = (unsigned)(uintptr_t)(const void*)&As[1][0];

    v8f acc[2][3] = {};
    const int nk = K / 32;

    // ---- prologue: TDM A(0), stage B(0)
    if (wv == 0) tdm_load_tile_64x32(abase, ldsA0, K);
    {
#pragma unroll
        for (int i = 0; i < 2; ++i) {
            int idx = tid + i * 192;           // 0..383
            int kk  = idx / 12;                // 0..31
            int nb  = (idx % 12) * 8;          // 0..88
            u128 r = *(const u128*)(Bw + (size_t)kk * N + n0 + nb);
            alignas(16) _Float16 t8[8];
            *(u128*)t8 = r;
#pragma unroll
            for (int j = 0; j < 8; ++j) Bs[0][(nb + j) * 48 + kk] = t8[j];
        }
    }

    const int alo = (lane < 16) ? 0 : 8;       // A runs {0..7,16..23} / {8..15,24..31}
    const int blo = (lane < 16) ? 0 : 8;       // B runs {0..7,16..23} / {8..15,24..31}

    for (int ks = 0; ks < nk; ++ks) {
        const int cur = ks & 1;
        if (wv == 0) __builtin_amdgcn_s_wait_tensorcnt(0);   // A(cur) in LDS
        __syncthreads();                                     // A+B(cur) visible to all

        // ---- prefetch next tile into alternate buffers
        if (ks + 1 < nk) {
            const int nxt = cur ^ 1;
            if (wv == 0)
                tdm_load_tile_64x32(abase + 2ull * (unsigned)((ks + 1) * 32),
                                    nxt ? ldsA1 : ldsA0, K);
            const int k0 = (ks + 1) * 32;
#pragma unroll
            for (int i = 0; i < 2; ++i) {
                int idx = tid + i * 192;
                int kk  = idx / 12;
                int nb  = (idx % 12) * 8;
                u128 r = *(const u128*)(Bw + (size_t)(k0 + kk) * N + n0 + nb);
                alignas(16) _Float16 t8[8];
                *(u128*)t8 = r;
#pragma unroll
                for (int j = 0; j < 8; ++j) Bs[nxt][(nb + j) * 48 + kk] = t8[j];
            }
        }

        // ---- fragments per ISA layout (16-bit A 16x32, B 32x16)
        AFrag a[2], bfr[3];
#pragma unroll
        for (int r = 0; r < 2; ++r) {
            const _Float16* base = &As[cur][(wrow + r * 16 + (lane & 15)) * 32];
            a[r].u[0] = *(const u128*)(base + alo);
            a[r].u[1] = *(const u128*)(base + alo + 16);
        }
#pragma unroll
        for (int c = 0; c < 3; ++c) {
            const _Float16* base = &Bs[cur][(wcol + c * 16 + (lane & 15)) * 48];
            bfr[c].u[0] = *(const u128*)(base + blo);
            bfr[c].u[1] = *(const u128*)(base + blo + 16);
        }
#pragma unroll
        for (int r = 0; r < 2; ++r)
#pragma unroll
            for (int c = 0; c < 3; ++c)
                acc[r][c] = __builtin_amdgcn_wmma_f32_16x16x32_f16(
                    false, a[r].v, false, bfr[c].v, (short)0, acc[r][c], false, false);
    }

    // ---- epilogue (C layout: lane&15 = N col; vgpr g = M row g (+8 for lanes>=16))
#pragma unroll
    for (int r = 0; r < 2; ++r) {
#pragma unroll
        for (int c = 0; c < 3; ++c) {
            int n = n0 + wcol + c * 16 + (lane & 15);
            float bv = bias[n];
#pragma unroll
            for (int g = 0; g < 8; ++g) {
                int m = m0 + wrow + r * 16 + g + ((lane >= 16) ? 8 : 0);
                float val = acc[r][c][g] + bv;
                size_t oi = (size_t)m * N + n;
                if (mode == 0) {
                    outH[oi] = (_Float16)val;
                } else if (mode == 2) {
                    float ge = 0.5f * val * (1.0f + erff(val * 0.70710678118654752f));
                    outH[oi] = (_Float16)ge;
                } else {
                    outF[oi] = val + resid[oi];
                }
            }
        }
    }
}

// ----------------------------------------------------------------------------
// Windowed attention: one wave per (batch, window, head).
// Computes S^T = K * Q^T via WMMA (softmax columns live in lanes l / l+16),
// adds rel-pos bias + shift mask, softmax, P round-trips through LDS
// transposed, then O = P * V via WMMA. Output scattered back to token order.
// ----------------------------------------------------------------------------
__device__ __forceinline__ int regid3(int c) { return c < 217 ? 0 : (c < 221 ? 1 : 2); }

__global__ __launch_bounds__(32)
void attn_kernel(const _Float16* __restrict__ qkv, const float* __restrict__ rpb,
                 _Float16* __restrict__ ao) {
    __shared__ __align__(16) _Float16 qs[64 * 32];   // [n][d]
    __shared__ __align__(16) _Float16 ks[64 * 32];   // [m][d]
    __shared__ __align__(16) _Float16 vT[32 * 64];   // [d][m]
    __shared__ __align__(16) _Float16 pT[16 * 64];   // [n_local][m]

    const int lane = threadIdx.x;
    const int bid  = blockIdx.x;
    const int h    = bid % NHEAD;
    const int wb   = bid / NHEAD;
    const int win  = wb % 1024;
    const int b    = wb / 1024;
    const int wy   = win >> 5, wx = win & 31;

    for (int i = lane; i < 64 * 32; i += 32) { qs[i] = (_Float16)0.f; ks[i] = (_Float16)0.f; }
    for (int i = lane; i < 32 * 64; i += 32) vT[i] = (_Float16)0.f;
    __syncthreads();

    // gather the 49 window tokens (shifted partition folded into addressing)
    for (int p = lane; p < 49; p += 32) {
        int ty = p / 7, tx = p % 7;
        int hh = wy * 7 + ty + 3; if (hh >= 224) hh -= 224;   // inverse roll(-3)
        int ww = wx * 7 + tx + 3; if (ww >= 224) ww -= 224;
        size_t tok = ((size_t)b * 224 + hh) * 224 + ww;
        const _Float16* base = qkv + tok * 576 + h * HDIM;
        const u128* gq = (const u128*)base;
        const u128* gk = (const u128*)(base + 192);
        const u128* gv = (const u128*)(base + 384);
        u128* sq = (u128*)(qs + p * 32);
        u128* sk = (u128*)(ks + p * 32);
#pragma unroll
        for (int j = 0; j < 4; ++j) { sq[j] = gq[j]; sk[j] = gk[j]; }
        alignas(16) _Float16 tv[32];
#pragma unroll
        for (int j = 0; j < 4; ++j) ((u128*)tv)[j] = gv[j];
#pragma unroll
        for (int d = 0; d < 32; ++d) vT[d * 64 + p] = tv[d];
    }
    __syncthreads();

    const int nl  = lane & 15;
    const int alo = (lane < 16) ? 0 : 8;
    const int blo = (lane < 16) ? 0 : 8;

    // K fragments (A-side of S^T = K * Q^T), reused across all query tiles
    AFrag ak[4];
#pragma unroll
    for (int mt = 0; mt < 4; ++mt) {
        const _Float16* base = ks + (mt * 16 + nl) * 32;
        ak[mt].u[0] = *(const u128*)(base + alo);
        ak[mt].u[1] = *(const u128*)(base + alo + 16);
    }

    for (int nt = 0; nt < 4; ++nt) {
        AFrag bq;
        {
            const _Float16* base = qs + (nt * 16 + nl) * 32;
            bq.u[0] = *(const u128*)(base + blo);
            bq.u[1] = *(const u128*)(base + blo + 16);
        }
        v8f st[4] = {};
#pragma unroll
        for (int mt = 0; mt < 4; ++mt)
            st[mt] = __builtin_amdgcn_wmma_f32_16x16x32_f16(
                false, ak[mt].v, false, bq.v, (short)0, st[mt], false, false);

        const int n = nt * 16 + nl;   // query index of this lane's column
#pragma unroll
        for (int mt = 0; mt < 4; ++mt) {
#pragma unroll
            for (int g = 0; g < 8; ++g) {
                int m = mt * 16 + g + ((lane >= 16) ? 8 : 0);   // key index
                float s;
                if (m >= 49) { s = -1e30f; }
                else if (n >= 49) { s = 0.0f; }
                else {
                    int in_ = n / 7, jn = n % 7, im = m / 7, jm = m % 7;
                    int idx = (in_ - im + 6) * 13 + (jn - jm + 6);
                    float bias = rpb[idx * NHEAD + h];
                    int rn = regid3(wy * 7 + in_) * 3 + regid3(wx * 7 + jn);
                    int rm = regid3(wy * 7 + im) * 3 + regid3(wx * 7 + jm);
                    float msk = (rn != rm) ? -100.0f : 0.0f;
                    s = st[mt][g] * 0.17677669529663687f + bias + msk;
                }
                st[mt][g] = s;
            }
        }
        // softmax over keys: in-lane (32 vals) + partner lane (l ^ 16)
        float mx = -1e30f;
#pragma unroll
        for (int mt = 0; mt < 4; ++mt)
#pragma unroll
            for (int g = 0; g < 8; ++g) mx = fmaxf(mx, st[mt][g]);
        mx = fmaxf(mx, __shfl_xor(mx, 16));
        float sum = 0.f;
#pragma unroll
        for (int mt = 0; mt < 4; ++mt)
#pragma unroll
            for (int g = 0; g < 8; ++g) { float e = __expf(st[mt][g] - mx); st[mt][g] = e; sum += e; }
        sum += __shfl_xor(sum, 16);
        float inv = 1.0f / (sum + 1e-20f);

        // write P^T to LDS so P can be re-read in A-fragment layout
#pragma unroll
        for (int mt = 0; mt < 4; ++mt)
#pragma unroll
            for (int g = 0; g < 8; ++g) {
                int m = mt * 16 + g + ((lane >= 16) ? 8 : 0);
                pT[nl * 64 + m] = (_Float16)(st[mt][g] * inv);
            }
        __syncthreads();

        // O(16xHD) = P(16x64) * V(64xHD): 2 k-steps of 32, 2 d-tiles
        AFrag ap[2];
#pragma unroll
        for (int k2 = 0; k2 < 2; ++k2) {
            const _Float16* base = pT + nl * 64 + k2 * 32;
            ap[k2].u[0] = *(const u128*)(base + alo);
            ap[k2].u[1] = *(const u128*)(base + alo + 16);
        }
        v8f oacc[2] = {};
#pragma unroll
        for (int dt = 0; dt < 2; ++dt) {
#pragma unroll
            for (int k2 = 0; k2 < 2; ++k2) {
                AFrag bv;
                const _Float16* base = vT + (dt * 16 + nl) * 64 + k2 * 32;
                bv.u[0] = *(const u128*)(base + blo);
                bv.u[1] = *(const u128*)(base + blo + 16);
                oacc[dt] = __builtin_amdgcn_wmma_f32_16x16x32_f16(
                    false, ap[k2].v, false, bv.v, (short)0, oacc[dt], false, false);
            }
        }
        // scatter O back to token order (pre-proj activation buffer)
#pragma unroll
        for (int dt = 0; dt < 2; ++dt) {
#pragma unroll
            for (int g = 0; g < 8; ++g) {
                int nn = nt * 16 + g + ((lane >= 16) ? 8 : 0);
                if (nn < 49) {
                    int ty = nn / 7, tx = nn % 7;
                    int hh = wy * 7 + ty + 3; if (hh >= 224) hh -= 224;
                    int ww = wx * 7 + tx + 3; if (ww >= 224) ww -= 224;
                    size_t tok = ((size_t)b * 224 + hh) * 224 + ww;
                    ao[tok * CDIM + h * HDIM + dt * 16 + nl] = (_Float16)oacc[dt][g];
                }
            }
        }
        __syncthreads();   // pT reused next iteration
    }
}

// ----------------------------------------------------------------------------
// Host-side orchestration
// ----------------------------------------------------------------------------
extern "C" void kernel_launch(void* const* d_in, const int* in_sizes, int n_in,
                              void* d_out, int out_size, void* d_ws, size_t ws_size,
                              hipStream_t stream) {
    const float* x      = (const float*)d_in[0];
    const float* qkv_w  = (const float*)d_in[3];
    const float* qkv_b  = (const float*)d_in[4];
    const float* proj_w = (const float*)d_in[5];
    const float* proj_b = (const float*)d_in[6];
    const float* rpb    = (const float*)d_in[7];
    const float* n1w    = (const float*)d_in[8];
    const float* n1b    = (const float*)d_in[9];
    const float* n2w    = (const float*)d_in[10];
    const float* n2b    = (const float*)d_in[11];
    const float* w1     = (const float*)d_in[12];
    const float* b1     = (const float*)d_in[13];
    const float* w2     = (const float*)d_in[14];
    const float* b2     = (const float*)d_in[15];
    (void)in_sizes; (void)n_in; (void)out_size; (void)ws_size;

    const size_t M = 401408;   // 8 * 224 * 224
    char* ws = (char*)d_ws;
    size_t off = 0;
    auto alloc = [&](size_t bytes) { size_t o = off; off += (bytes + 255) & ~(size_t)255; return o; };
    _Float16* XN  = (_Float16*)(ws + alloc(M * 192 * 2));   // LN1 out, reused for LN2 out
    _Float16* QKV = (_Float16*)(ws + alloc(M * 768 * 2));   // QKV (576 cols), reused as MLP hidden (768)
    _Float16* AO  = (_Float16*)(ws + alloc(M * 192 * 2));   // attention out (pre-proj)
    float*    X2  = (float*)   (ws + alloc(M * 192 * 4));   // x + attn(proj) residual
    _Float16* QW  = (_Float16*)(ws + alloc(192 * 576 * 2));
    _Float16* PW  = (_Float16*)(ws + alloc(192 * 192 * 2));
    _Float16* W1h = (_Float16*)(ws + alloc(192 * 768 * 2));
    _Float16* W2h = (_Float16*)(ws + alloc(768 * 192 * 2));

    cvt_f16_kernel<<<(110592 + 255) / 256, 256, 0, stream>>>(qkv_w,  QW,  110592);
    cvt_f16_kernel<<<(36864  + 255) / 256, 256, 0, stream>>>(proj_w, PW,  36864);
    cvt_f16_kernel<<<(147456 + 255) / 256, 256, 0, stream>>>(w1,     W1h, 147456);
    cvt_f16_kernel<<<(147456 + 255) / 256, 256, 0, stream>>>(w2,     W2h, 147456);

    // LN1
    ln_kernel<<<M / 8, 256, 0, stream>>>(x, n1w, n1b, XN, (int)M);
    // QKV projection (position independent -> flat GEMM)
    gemm_wmma_kernel<<<dim3(576 / 96, M / 64), 192, 0, stream>>>(
        XN, QW, qkv_b, nullptr, nullptr, QKV, (int)M, 192, 576, 0);
    // shifted-window attention (gather/scatter folds roll + window partition)
    attn_kernel<<<8 * 1024 * 6, 32, 0, stream>>>(QKV, rpb, AO);
    // proj + shortcut -> x2
    gemm_wmma_kernel<<<dim3(192 / 96, M / 64), 192, 0, stream>>>(
        AO, PW, proj_b, x, X2, nullptr, (int)M, 192, 192, 1);
    // LN2
    ln_kernel<<<M / 8, 256, 0, stream>>>(X2, n2w, n2b, XN, (int)M);
    // MLP1 + exact GELU
    gemm_wmma_kernel<<<dim3(768 / 96, M / 64), 192, 0, stream>>>(
        XN, W1h, b1, nullptr, nullptr, QKV, (int)M, 192, 768, 2);
    // MLP2 + residual -> final output
    gemm_wmma_kernel<<<dim3(192 / 96, M / 64), 192, 0, stream>>>(
        QKV, W2h, b2, X2, (float*)d_out, nullptr, (int)M, 768, 192, 1);
}
